// DeepseekV4MoE_61718680043942
// MI455X (gfx1250) — matrix operand order repaired and checked by reading, not compiled, not run on hardware
//
#include <hip/hip_runtime.h>
#include <hip/hip_bf16.h>

// ---------------- problem constants (match reference) ----------------
constexpr int T  = 1024;   // tokens
constexpr int H  = 1024;   // hidden
constexpr int E  = 16;     // experts
constexpr int I  = 512;    // expert ffn dim
constexpr int K  = 4;      // top-k
constexpr int IS = 1024;   // shared intermediate
constexpr float LIMIT = 7.0f;
constexpr float SCALE = 2.5f;

typedef __attribute__((ext_vector_type(16))) __bf16 v16bf;
typedef __attribute__((ext_vector_type(8)))  float  v8f;

// ---------------- WMMA helpers (CDNA5, wave32) ----------------
// D = A(16x32 bf16) * B(32x16 bf16) + C(16x16 f32)
__device__ __forceinline__ v8f wmma_bf16(v16bf a, v16bf b, v8f c) {
  return __builtin_amdgcn_wmma_f32_16x16x32_bf16(
      /*neg_a=*/false, a, /*neg_b=*/false, b,
      /*c_mod=*/(short)0, c, /*reuse_a=*/false, /*reuse_b=*/false);
}

// fast sigmoid: single v_rcp_f32 instead of IEEE division sequence
__device__ __forceinline__ float fast_sigmoid(float z) {
  return __builtin_amdgcn_rcpf(1.0f + __expf(-z));
}

// A-matrix 16x32 bf16 per ISA 7.12.2: lane L holds row M=L&15;
// half (L>>4) selects K offsets {kb..kb+7} (V0..V3) and {16+kb..+7} (V4..V7)
// -> two contiguous 16B loads from a bf16 row.
__device__ __forceinline__ v16bf load_a_bf16(const __bf16* row, int lane) {
  int kb = (lane >> 4) * 8;
  union { uint4 q[2]; v16bf v; } u;
  u.q[0] = *(const uint4*)(row + kb);
  u.q[1] = *(const uint4*)(row + 16 + kb);
  return u.v;
}

// B fragments are pre-swizzled: tile (n_tile, k_tile) stored as 32 lanes x 16 bf16,
// 1024B per tile, k_tile-major within a column tile run.
// Per-lane read = 32 contiguous bytes = 2x global_load_b128, fully coalesced.
__device__ __forceinline__ v16bf load_b_frag(const __bf16* FB, int Kt,
                                             int k_tile, int n_tile, int lane) {
  const __bf16* p = FB + ((size_t)(n_tile * Kt + k_tile) << 9) + lane * 16;
  union { uint4 q[2]; v16bf v; } u;
  u.q[0] = *(const uint4*)(p);
  u.q[1] = *(const uint4*)(p + 8);
  return u.v;
}

// ---------------- double-buffered (ping-pong) WMMA GEMM inner loop ----------------
// While WMMAs consume one buffer, the next K-tile's loads fill the other, so the
// per-WMMA s_wait_loadcnt leaves the next tile's loads in flight.
// KT is a compile-time power of two: the prefetch index wraps with a mask
// (final prefetch is a redundant in-bounds reload, no select chain).
template <int NB, int KT>
__device__ __forceinline__ void gemm_pipe(const __bf16* arow,
                                          const __bf16* const (&Fb)[NB],
                                          const int (&nt)[NB],
                                          int lane, v8f (&acc)[NB]) {
  static_assert((KT & (KT - 1)) == 0 && KT >= 2, "KT must be a power of two");
  v16bf a0 = load_a_bf16(arow, lane);
  v16bf b0[NB];
#pragma unroll
  for (int i = 0; i < NB; ++i) b0[i] = load_b_frag(Fb[i], KT, 0, nt[i], lane);
  for (int kt = 0; kt < KT; kt += 2) {
    int k1 = kt + 1;
    v16bf a1 = load_a_bf16(arow + k1 * 32, lane);
    v16bf b1[NB];
#pragma unroll
    for (int i = 0; i < NB; ++i) b1[i] = load_b_frag(Fb[i], KT, k1, nt[i], lane);
#pragma unroll
    for (int i = 0; i < NB; ++i) acc[i] = wmma_bf16(a0, b0[i], acc[i]);
    int k2 = (kt + 2) & (KT - 1);           // wraps to 0 on last (redundant) prefetch
    a0 = load_a_bf16(arow + k2 * 32, lane);
#pragma unroll
    for (int i = 0; i < NB; ++i) b0[i] = load_b_frag(Fb[i], KT, k2, nt[i], lane);
#pragma unroll
    for (int i = 0; i < NB; ++i) acc[i] = wmma_bf16(a1, b1[i], acc[i]);
  }
}

// ---------------- weight fragmentizer: fp32 [Kdim x Ndim] -> bf16 B-fragments ----
__global__ __launch_bounds__(256) void fragmentize_kernel(
    const float* __restrict__ W, __bf16* __restrict__ FB,
    int Kdim, int Ndim, size_t srcStride, size_t dstStride) {
  const float* Ws = W + (size_t)blockIdx.y * srcStride;
  __bf16* FBs = FB + (size_t)blockIdx.y * dstStride;
  int Kt = Kdim >> 5, Nt = Ndim >> 4;
  int lane = threadIdx.x & 31, wave = threadIdx.x >> 5;
  int tile = blockIdx.x * 8 + wave;
  if (tile >= Kt * Nt) return;
  int n_tile = tile / Kt, k_tile = tile - n_tile * Kt;
  int n  = n_tile * 16 + (lane & 15);
  int kb = k_tile * 32 + ((lane >> 4) << 3);
  const float* src = Ws + (size_t)kb * Ndim + n;
  __bf16* dst = FBs + ((size_t)tile << 9) + lane * 16;
  union { uint4 q[2]; __bf16 h[16]; } u;
#pragma unroll
  for (int j = 0; j < 16; ++j) u.h[j] = (__bf16)src[(size_t)(j < 8 ? j : j + 8) * Ndim];
  *(uint4*)dst = u.q[0];
  *(uint4*)(dst + 8) = u.q[1];
}

// ---------------- zero expert counters ----------------
__global__ void zero_cnt_kernel(int* cnt) {
  if (threadIdx.x < E) cnt[threadIdx.x] = 0;
}

// ---------------- router (one block per token) ----------------
__global__ __launch_bounds__(256) void router_kernel(
    const float* __restrict__ x, const float* __restrict__ Wg,
    const float* __restrict__ bias, int* __restrict__ cnt,
    int* __restrict__ slots, float* __restrict__ wts) {
  int t = blockIdx.x;
  int tid = threadIdx.x;
  int e = tid & 15, g = tid >> 4;
  const float* xr = x + (size_t)t * H;
  const float* wr = Wg + (size_t)e * H;
  float acc = 0.f;
  for (int h = g; h < H; h += 16) acc += xr[h] * wr[h];
  __shared__ float red[256];
  red[tid] = acc;
  __syncthreads();
#pragma unroll
  for (int s = 128; s >= 16; s >>= 1) {
    if (tid < s) red[tid] += red[tid + s];
    __syncthreads();
  }
  __shared__ float sc[E], scb[E];
  if (tid < E) {
    float l = red[tid];
    float sp = (l > 20.f) ? l : log1pf(__expf(l));  // stable softplus
    float s = sqrtf(sp);                             // sqrtsoftplus score
    sc[tid]  = s;
    scb[tid] = s + bias[tid];                        // biased selection score
  }
  __syncthreads();
  if (tid == 0) {
    unsigned mask = 0;
    int sel[K]; float wv[K]; float wsum = 0.f;
#pragma unroll
    for (int k = 0; k < K; ++k) {
      int best = 0; float bv = -1e30f;
      for (int ee = 0; ee < E; ++ee)
        if (!((mask >> ee) & 1u) && scb[ee] > bv) { bv = scb[ee]; best = ee; }
      mask |= 1u << best;
      sel[k] = best; wv[k] = sc[best]; wsum += sc[best];
    }
    float inv = SCALE / wsum;
#pragma unroll
    for (int k = 0; k < K; ++k) {
      int ee = sel[k];
      int pos = atomicAdd(&cnt[ee], 1);
      slots[ee * T + pos] = t * K + k;   // global (token,k) slot id
      wts[ee * T + pos]   = wv[k] * inv;
    }
  }
}

// ---------------- x (f32) -> xb (bf16) ----------------
__global__ void cvt_kernel(const float* __restrict__ x, __bf16* __restrict__ xb) {
  int i = (blockIdx.x * 256 + threadIdx.x) * 4;
  float4 v = *(const float4*)(x + i);
  xb[i + 0] = (__bf16)v.x;
  xb[i + 1] = (__bf16)v.y;
  xb[i + 2] = (__bf16)v.z;
  xb[i + 3] = (__bf16)v.w;
}

// ---------------- shared experts gate/up ----------------
// grid (T/16, IS/512), block 256 = 8 waves x 64 cols; s1 = silu(x@Wsg)*(x@Wsu)
__global__ __launch_bounds__(256) void shared_up_kernel(
    const __bf16* __restrict__ xb, const __bf16* __restrict__ Fg,
    const __bf16* __restrict__ Fu, __bf16* __restrict__ s1) {
  int t0   = blockIdx.x * 16;
  int lane = threadIdx.x & 31;
  int wave = threadIdx.x >> 5;
  int nt0  = blockIdx.y * 32 + wave * 4;       // base 16-col tile index
  const __bf16* arow = xb + (size_t)(t0 + (lane & 15)) * H;
  const __bf16* Fb[8] = {Fg, Fg, Fg, Fg, Fu, Fu, Fu, Fu};
  const int nt[8] = {nt0, nt0 + 1, nt0 + 2, nt0 + 3,
                     nt0, nt0 + 1, nt0 + 2, nt0 + 3};
  v8f acc[8] = {};
  gemm_pipe<8, H / 32>(arow, Fb, nt, lane, acc);
  int mbase = (lane >> 4) << 3, col = lane & 15;
#pragma unroll
  for (int ct = 0; ct < 4; ++ct)
#pragma unroll
    for (int r = 0; r < 8; ++r) {
      float g = acc[ct][r];
      float u = acc[ct + 4][r];
      float v = g * fast_sigmoid(g) * u;          // silu(g)*u
      s1[(size_t)(t0 + mbase + r) * IS + (nt0 + ct) * 16 + col] = (__bf16)v;
    }
}

// ---------------- shared experts down (initializes out) ----------------
// grid (T/16), block 256 = 8 waves x 128 cols = full H
__global__ __launch_bounds__(256) void shared_down_kernel(
    const __bf16* __restrict__ s1, const __bf16* __restrict__ Fd,
    float* __restrict__ out) {
  int t0   = blockIdx.x * 16;
  int lane = threadIdx.x & 31;
  int nt0  = (threadIdx.x >> 5) * 8;
  const __bf16* arow = s1 + (size_t)(t0 + (lane & 15)) * IS;
  const __bf16* Fb[8] = {Fd, Fd, Fd, Fd, Fd, Fd, Fd, Fd};
  const int nt[8] = {nt0, nt0 + 1, nt0 + 2, nt0 + 3,
                     nt0 + 4, nt0 + 5, nt0 + 6, nt0 + 7};
  v8f acc[8] = {};
  gemm_pipe<8, IS / 32>(arow, Fb, nt, lane, acc);
  int mbase = (lane >> 4) << 3, col = lane & 15;
#pragma unroll
  for (int ct = 0; ct < 8; ++ct)
#pragma unroll
    for (int r = 0; r < 8; ++r)
      out[(size_t)(t0 + mbase + r) * H + (nt0 + ct) * 16 + col] = acc[ct][r];
}

// ---------------- routed experts gate/up + clamped swiglu ----------------
// grid (E, T/16 max tiles), block 256 = 8 waves x 64 cols = full I
__global__ __launch_bounds__(256) void moe_act_kernel(
    const __bf16* __restrict__ xb, const __bf16* __restrict__ F1,
    const __bf16* __restrict__ F3, const int* __restrict__ cnt,
    const int* __restrict__ slots, __bf16* __restrict__ act) {
  int e = blockIdx.x, tile = blockIdx.y;
  int n = cnt[e];
  if (tile * 16 >= n) return;
  int lane = threadIdx.x & 31;
  int wave = threadIdx.x >> 5;
  const __bf16* F1e = F1 + (size_t)e * H * I;
  const __bf16* F3e = F3 + (size_t)e * H * I;
  __shared__ int s_slot[16];
  if (threadIdx.x < 16) {
    int idx = tile * 16 + threadIdx.x;
    if (idx >= n) idx = n - 1;                 // padded rows duplicate last entry
    s_slot[threadIdx.x] = slots[e * T + idx];
  }
  __syncthreads();
  const __bf16* arow = xb + (size_t)(s_slot[lane & 15] >> 2) * H;  // token = slot>>2
  int nt0 = wave * 4;
  const __bf16* Fb[8] = {F1e, F1e, F1e, F1e, F3e, F3e, F3e, F3e};
  const int nt[8] = {nt0, nt0 + 1, nt0 + 2, nt0 + 3,
                     nt0, nt0 + 1, nt0 + 2, nt0 + 3};
  v8f acc[8] = {};
  gemm_pipe<8, H / 32>(arow, Fb, nt, lane, acc);
  int mbase = (lane >> 4) << 3, col = lane & 15;
#pragma unroll
  for (int ct = 0; ct < 4; ++ct)
#pragma unroll
    for (int r = 0; r < 8; ++r) {
      float g = fminf(acc[ct][r], LIMIT);
      float u = fminf(fmaxf(acc[ct + 4][r], -LIMIT), LIMIT);
      float v = g * fast_sigmoid(1.702f * g) * (u + 1.0f);   // clamped swiglu
      int s = s_slot[mbase + r];
      act[(size_t)s * I + (nt0 + ct) * 16 + col] = (__bf16)v;
    }
}

// ---------------- routed down + weighted atomic combine ----------------
// grid (E, T/16), block 256 = 8 waves x 128 cols = full H
__global__ __launch_bounds__(256) void moe_down_kernel(
    const __bf16* __restrict__ act, const __bf16* __restrict__ F2,
    const int* __restrict__ cnt, const int* __restrict__ slots,
    const float* __restrict__ wts, float* __restrict__ out) {
  int e = blockIdx.x, tile = blockIdx.y;
  int n = cnt[e];
  if (tile * 16 >= n) return;
  int lane = threadIdx.x & 31;
  int wave = threadIdx.x >> 5;
  const __bf16* F2e = F2 + (size_t)e * I * H;
  __shared__ int   s_slot[16];
  __shared__ float s_w[16];
  if (threadIdx.x < 16) {
    int idx = tile * 16 + threadIdx.x;
    float w = 0.f;
    if (idx >= n) idx = n - 1; else w = wts[e * T + idx];   // pads get weight 0
    s_slot[threadIdx.x] = slots[e * T + idx];
    s_w[threadIdx.x]    = w;
  }
  __syncthreads();
  const __bf16* arow = act + (size_t)s_slot[lane & 15] * I;
  int nt0 = wave * 8;
  const __bf16* Fb[8] = {F2e, F2e, F2e, F2e, F2e, F2e, F2e, F2e};
  const int nt[8] = {nt0, nt0 + 1, nt0 + 2, nt0 + 3,
                     nt0 + 4, nt0 + 5, nt0 + 6, nt0 + 7};
  v8f acc[8] = {};
  gemm_pipe<8, I / 32>(arow, Fb, nt, lane, acc);
  int mbase = (lane >> 4) << 3, col = lane & 15;
#pragma unroll
  for (int ct = 0; ct < 8; ++ct)
#pragma unroll
    for (int r = 0; r < 8; ++r) {
      int mm = mbase + r;
      int t  = s_slot[mm] >> 2;
      float v = acc[ct][r] * s_w[mm];
      atomicAdd(&out[(size_t)t * H + (nt0 + ct) * 16 + col], v);
    }
}

// ---------------- host launcher ----------------
extern "C" void kernel_launch(void* const* d_in, const int* in_sizes, int n_in,
                              void* d_out, int out_size, void* d_ws, size_t ws_size,
                              hipStream_t stream) {
  (void)in_sizes; (void)n_in; (void)out_size; (void)ws_size;
  const float* x    = (const float*)d_in[0];
  // d_in[1] = input_ids (unused by the reference computation)
  const float* Wg   = (const float*)d_in[2];
  const float* bias = (const float*)d_in[3];
  const float* W1   = (const float*)d_in[4];
  const float* W3   = (const float*)d_in[5];
  const float* W2   = (const float*)d_in[6];
  const float* Wsg  = (const float*)d_in[7];
  const float* Wsu  = (const float*)d_in[8];
  const float* Wsd  = (const float*)d_in[9];
  float* out = (float*)d_out;

  // workspace carve-up (all offsets 256B aligned); total ~62 MB
  char* w = (char*)d_ws;
  int*    cnt   = (int*)w;            w += 256;
  int*    slots = (int*)w;            w += (size_t)E * T * sizeof(int);
  float*  wts   = (float*)w;          w += (size_t)E * T * sizeof(float);
  __bf16* xb    = (__bf16*)w;         w += (size_t)T * H * sizeof(__bf16);
  __bf16* actb  = (__bf16*)w;         w += (size_t)T * K * I * sizeof(__bf16);
  __bf16* s1    = (__bf16*)w;         w += (size_t)T * IS * sizeof(__bf16);
  __bf16* F1    = (__bf16*)w;         w += (size_t)E * H * I * sizeof(__bf16);
  __bf16* F3    = (__bf16*)w;         w += (size_t)E * H * I * sizeof(__bf16);
  __bf16* F2    = (__bf16*)w;         w += (size_t)E * I * H * sizeof(__bf16);
  __bf16* Fg    = (__bf16*)w;         w += (size_t)H * IS * sizeof(__bf16);
  __bf16* Fu    = (__bf16*)w;         w += (size_t)H * IS * sizeof(__bf16);
  __bf16* Fd    = (__bf16*)w;         w += (size_t)IS * H * sizeof(__bf16);

  zero_cnt_kernel<<<1, 32, 0, stream>>>(cnt);
  router_kernel<<<T, 256, 0, stream>>>(x, Wg, bias, cnt, slots, wts);
  cvt_kernel<<<(T * H) / 1024, 256, 0, stream>>>(x, xb);

  // one-time bf16 B-fragment conversion (reused ~64x by the GEMMs below from L2)
  fragmentize_kernel<<<dim3(128, E), 256, 0, stream>>>(W1, F1, H, I,
      (size_t)H * I, (size_t)H * I);
  fragmentize_kernel<<<dim3(128, E), 256, 0, stream>>>(W3, F3, H, I,
      (size_t)H * I, (size_t)H * I);
  fragmentize_kernel<<<dim3(128, E), 256, 0, stream>>>(W2, F2, I, H,
      (size_t)I * H, (size_t)I * H);
  fragmentize_kernel<<<dim3(256, 1), 256, 0, stream>>>(Wsg, Fg, H, IS, 0, 0);
  fragmentize_kernel<<<dim3(256, 1), 256, 0, stream>>>(Wsu, Fu, H, IS, 0, 0);
  fragmentize_kernel<<<dim3(256, 1), 256, 0, stream>>>(Wsd, Fd, IS, H, 0, 0);

  shared_up_kernel<<<dim3(T / 16, IS / 512), 256, 0, stream>>>(xb, Fg, Fu, s1);
  shared_down_kernel<<<T / 16, 256, 0, stream>>>(s1, Fd, out);      // writes out
  moe_act_kernel<<<dim3(E, T / 16), 256, 0, stream>>>(xb, F1, F3, cnt, slots, actb);
  moe_down_kernel<<<dim3(E, T / 16), 256, 0, stream>>>(actb, F2, cnt, slots, wts, out); // += routed
}
